// UniversalGlobalAttention_24515673326271
// MI455X (gfx1250) — matrix operand order repaired and checked by reading, not compiled, not run on hardware
//
#include <hip/hip_runtime.h>
#include <hip/hip_bf16.h>
#include <math.h>

// ---------------- problem constants ----------------
#define NSRC   100000
#define NDST   100000
#define HID    128
#define NHEADS 8
#define HDIM   16
#define NEDGES 640000
#define SCALING 0.25f   // HDIM^-0.5 = 16^-0.5

// LDS weight tile: 128 rows x 128 cols f16, rows padded by 8 halves (16B)
// so consecutive rows start 68 banks apart -> fragment reads spread banks.
#define LW_STRIDE 136   // halves per row

typedef __attribute__((ext_vector_type(16))) _Float16 v16h;
typedef __attribute__((ext_vector_type(8)))  _Float16 v8h;
typedef __attribute__((ext_vector_type(4)))  _Float16 v4h;
typedef __attribute__((ext_vector_type(8)))  float    v8f;

// ---------------- helpers ----------------
// order-preserving float -> uint encoding for exact atomic max on floats
__device__ __forceinline__ unsigned encF(float f) {
    unsigned u = __float_as_uint(f);
    return (u & 0x80000000u) ? ~u : (u | 0x80000000u);
}
__device__ __forceinline__ float decF(unsigned e) {
    return (e & 0x80000000u) ? __uint_as_float(e ^ 0x80000000u)
                             : __uint_as_float(~e);
}

// load 2x8 contiguous f32 from global and pack to 16 f16 (A fragment)
__device__ __forceinline__ v16h ld16h(const float* __restrict__ p0,
                                      const float* __restrict__ p1) {
    v16h r;
#pragma unroll
    for (int i = 0; i < 8; ++i) r[i]     = (_Float16)p0[i];
#pragma unroll
    for (int i = 0; i < 8; ++i) r[8 + i] = (_Float16)p1[i];
    return r;
}
__device__ __forceinline__ v16h ld16h_scaled(const float* __restrict__ p0, float s0,
                                             const float* __restrict__ p1, float s1) {
    v16h r;
#pragma unroll
    for (int i = 0; i < 8; ++i) r[i]     = (_Float16)(p0[i] * s0);
#pragma unroll
    for (int i = 0; i < 8; ++i) r[8 + i] = (_Float16)(p1[i] * s1);
    return r;
}

// 16 contiguous f16 from LDS (two 16B chunks; 16B aligned by construction)
__device__ __forceinline__ v16h ld16h_lds(const _Float16* p) {
    v8h a = *(const v8h*)p;
    v8h b = *(const v8h*)(p + 8);
    v16h r;
#pragma unroll
    for (int i = 0; i < 8; ++i) { r[i] = a[i]; r[8 + i] = b[i]; }
    return r;
}

// cooperative: stage 128x128 f32 weight matrix into LDS as f16 (padded rows)
__device__ __forceinline__ void stage_weights(const float* __restrict__ W,
                                              _Float16* lw) {
    const float4* w4 = (const float4*)W;
#pragma unroll
    for (int i = 0; i < 16; ++i) {                 // 4096 float4 / 256 threads
        int idx = i * 256 + (int)threadIdx.x;      // float4 index
        int row = idx >> 5;                        // 32 float4 per row
        int c4  = idx & 31;
        float4 wv = w4[idx];
        v4h p;
        p[0] = (_Float16)wv.x; p[1] = (_Float16)wv.y;
        p[2] = (_Float16)wv.z; p[3] = (_Float16)wv.w;
        *(v4h*)(lw + row * LW_STRIDE + c4 * 4) = p;  // 8B-aligned ds_store
    }
}

// ---------------- kernel 0: workspace init ----------------
__global__ void init_ws_kernel(unsigned* __restrict__ mEnc,
                               float* __restrict__ zsum,
                               float* __restrict__ agg) {
    const size_t stride = (size_t)gridDim.x * blockDim.x;
    size_t i = (size_t)blockIdx.x * blockDim.x + threadIdx.x;
    for (size_t j = i; j < (size_t)NSRC * NHEADS; j += stride) {
        mEnc[j] = 0x007FFFFFu;  // enc(-inf)
        zsum[j] = 0.0f;
    }
    for (size_t j = i; j < (size_t)NSRC * HID; j += stride) agg[j] = 0.0f;
}

// ---------------- kernel 1: Y[M,128] = X[M,128] @ W^T + b  (WMMA f16->f32) ----
// block = 8 waves; W staged once per block in LDS; one wave -> 16 rows x 128 cols
__global__ void gemm_proj_kernel(const float* __restrict__ X,
                                 const float* __restrict__ W,
                                 const float* __restrict__ bias,
                                 float* __restrict__ Y, int M) {
    __shared__ _Float16 lw[128 * LW_STRIDE];
    stage_weights(W, lw);
    __syncthreads();

    const int lane = threadIdx.x & 31;
    const int wave = threadIdx.x >> 5;
    const int tile = blockIdx.x * (blockDim.x >> 5) + wave;
    if (tile * 16 < M) {
        const int  row0 = tile * 16;
        const int  lrow = lane & 15;
        const bool hi   = lane >= 16;
        const int  koffA = hi ? 8  : 0;   // unused: A is also read at koffB (16-element K chunk)
        const int  koffB = hi ? 16 : 0;   // B: 16-element K chunk per half-wave

        v8f acc[8];
#pragma unroll
        for (int t = 0; t < 8; ++t)
#pragma unroll
            for (int r = 0; r < 8; ++r) acc[t][r] = 0.0f;

        const float* xrow = X + (size_t)(row0 + lrow) * HID;
#pragma unroll
        for (int kb = 0; kb < HID; kb += 32) {
            v16h a = ld16h(xrow + kb + koffB, xrow + kb + 8 + koffB);
#pragma unroll
            for (int t = 0; t < 8; ++t) {
                v16h b = ld16h_lds(&lw[(t * 16 + lrow) * LW_STRIDE + kb + koffB]);
                acc[t] = __builtin_amdgcn_wmma_f32_16x16x32_f16(
                    false, a, false, b, (short)0, acc[t], false, false);
            }
        }
#pragma unroll
        for (int t = 0; t < 8; ++t) {
            const int   col = t * 16 + lrow;
            const float bv  = bias[col];
#pragma unroll
            for (int r = 0; r < 8; ++r)
                Y[(size_t)(row0 + r + (hi ? 8 : 0)) * HID + col] = acc[t][r] + bv;
        }
    }
}

// ---------------- kernel 2: per-edge scores + segment max ----------------
__global__ void edge_score_max_kernel(const float* __restrict__ Q,
                                      const float* __restrict__ K,
                                      const int* __restrict__ sidx,
                                      const int* __restrict__ didx,
                                      float* __restrict__ scores,
                                      unsigned* __restrict__ mEnc) {
    const int t = blockIdx.x * blockDim.x + threadIdx.x;
    if (t >= NEDGES * NHEADS) return;
    const int e = t >> 3, h = t & 7;
    const int s = sidx[e], d = didx[e];
    const float4* q = (const float4*)(Q + (size_t)s * HID + h * HDIM);
    const float4* k = (const float4*)(K + (size_t)d * HID + h * HDIM);
    float acc = 0.0f;
#pragma unroll
    for (int i = 0; i < 4; ++i) {
        float4 a = q[i], b = k[i];
        acc += a.x * b.x + a.y * b.y + a.z * b.z + a.w * b.w;
    }
    acc *= SCALING;
    scores[t] = acc;
    atomicMax(&mEnc[s * NHEADS + h], encF(acc));
}

// ---------------- kernel 3: exp / z-sum / weighted scatter of V -------------
__global__ void edge_softmax_agg_kernel(const float* __restrict__ V,
                                        const int* __restrict__ sidx,
                                        const int* __restrict__ didx,
                                        const float* __restrict__ scores,
                                        const unsigned* __restrict__ mEnc,
                                        float* __restrict__ zsum,
                                        float* __restrict__ agg) {
    const int t = blockIdx.x * blockDim.x + threadIdx.x;
    if (t >= NEDGES * NHEADS) return;
    const int e = t >> 3, h = t & 7;
    const int s = sidx[e], d = didx[e];
    const float m = decF(mEnc[s * NHEADS + h]);
    const float w = expf(scores[t] - m);
    atomicAdd(&zsum[s * NHEADS + h], w);
    const float* vp = V + (size_t)d * HID + h * HDIM;
    float*       ap = agg + (size_t)s * HID + h * HDIM;
#pragma unroll
    for (int i = 0; i < HDIM; ++i) atomicAdd(&ap[i], w * vp[i]);
}

// ---------------- kernel 4: out = (agg / z) @ Wo^T + bo  (fused normalize) --
__global__ void gemm_out_kernel(const float* __restrict__ agg,
                                const float* __restrict__ zsum,
                                const float* __restrict__ W,
                                const float* __restrict__ bias,
                                float* __restrict__ Y, int M) {
    __shared__ _Float16 lw[128 * LW_STRIDE];
    stage_weights(W, lw);
    __syncthreads();

    const int lane = threadIdx.x & 31;
    const int wave = threadIdx.x >> 5;
    const int tile = blockIdx.x * (blockDim.x >> 5) + wave;
    if (tile * 16 < M) {
        const int  row0 = tile * 16;
        const int  lrow = lane & 15;
        const bool hi   = lane >= 16;
        const int  koffA = hi ? 8  : 0;
        const int  koffB = hi ? 16 : 0;

        // per-row reciprocal z per head (empty segments -> 0, as in reference)
        float rz[NHEADS];
        {
            const float* zr = zsum + (size_t)(row0 + lrow) * NHEADS;
#pragma unroll
            for (int h = 0; h < NHEADS; ++h) {
                float z = zr[h];
                rz[h] = (z > 0.0f) ? (1.0f / z) : 0.0f;
            }
        }

        v8f acc[8];
#pragma unroll
        for (int t = 0; t < 8; ++t)
#pragma unroll
            for (int r = 0; r < 8; ++r) acc[t][r] = 0.0f;

        const float* xrow = agg + (size_t)(row0 + lrow) * HID;
#pragma unroll
        for (int kb = 0; kb < HID; kb += 32) {
            const int h0 = (kb + koffB) >> 4;       // 8-chunk lies inside one head
            const int h1 = (kb + 8 + koffB) >> 4;
            v16h a = ld16h_scaled(xrow + kb + koffB, rz[h0],
                                  xrow + kb + 8 + koffB, rz[h1]);
#pragma unroll
            for (int t = 0; t < 8; ++t) {
                v16h b = ld16h_lds(&lw[(t * 16 + lrow) * LW_STRIDE + kb + koffB]);
                acc[t] = __builtin_amdgcn_wmma_f32_16x16x32_f16(
                    false, a, false, b, (short)0, acc[t], false, false);
            }
        }
#pragma unroll
        for (int t = 0; t < 8; ++t) {
            const int   col = t * 16 + lrow;
            const float bv  = bias[col];
#pragma unroll
            for (int r = 0; r < 8; ++r)
                Y[(size_t)(row0 + r + (hi ? 8 : 0)) * HID + col] = acc[t][r] + bv;
        }
    }
}

// ---------------- launcher ----------------
extern "C" void kernel_launch(void* const* d_in, const int* in_sizes, int n_in,
                              void* d_out, int out_size, void* d_ws, size_t ws_size,
                              hipStream_t stream) {
    (void)in_sizes; (void)n_in; (void)out_size; (void)ws_size;

    const float* src_x = (const float*)d_in[0];
    const float* dst_x = (const float*)d_in[1];
    const float* Wq = (const float*)d_in[2];
    const float* bq = (const float*)d_in[3];
    const float* Wk = (const float*)d_in[4];
    const float* bk = (const float*)d_in[5];
    const float* Wv = (const float*)d_in[6];
    const float* bv = (const float*)d_in[7];
    const float* Wo = (const float*)d_in[8];
    const float* bo = (const float*)d_in[9];
    const int*   sidx = (const int*)d_in[10];            // edge_index[0]
    const int*   didx = (const int*)d_in[10] + NEDGES;   // edge_index[1]
    float* out = (float*)d_out;

    // workspace layout (all f32-sized slots)
    float*    ws     = (float*)d_ws;
    float*    Q      = ws;                                   // NSRC*128
    float*    Kp     = Q  + (size_t)NSRC * HID;              // NDST*128
    float*    Vp     = Kp + (size_t)NDST * HID;              // NDST*128
    float*    scores = Vp + (size_t)NDST * HID;              // NEDGES*8
    unsigned* mEnc   = (unsigned*)(scores + (size_t)NEDGES * NHEADS); // NSRC*8
    float*    zsum   = (float*)mEnc + (size_t)NSRC * NHEADS; // NSRC*8
    float*    agg    = zsum + (size_t)NSRC * NHEADS;         // NSRC*128

    // 0) init accumulators
    init_ws_kernel<<<2048, 256, 0, stream>>>(mEnc, zsum, agg);

    // 1) node projections via WMMA: Q = src_x@Wq^T+bq, K/V from dst_x
    const int tilesS = NSRC / 16;                // 6250
    const int gridS  = (tilesS + 7) / 8;         // 8 waves / block
    gemm_proj_kernel<<<gridS, 256, 0, stream>>>(src_x, Wq, bq, Q,  NSRC);
    gemm_proj_kernel<<<gridS, 256, 0, stream>>>(dst_x, Wk, bk, Kp, NDST);
    gemm_proj_kernel<<<gridS, 256, 0, stream>>>(dst_x, Wv, bv, Vp, NDST);

    // 2) per-edge scores + segment max (one thread per edge-head)
    const int nEH  = NEDGES * NHEADS;            // 5.12M
    const int gEH  = (nEH + 255) / 256;
    edge_score_max_kernel<<<gEH, 256, 0, stream>>>(Q, Kp, sidx, didx, scores, mEnc);

    // 3) exp, z-sum, weighted scatter of V
    edge_softmax_agg_kernel<<<gEH, 256, 0, stream>>>(Vp, sidx, didx, scores,
                                                     mEnc, zsum, agg);

    // 4) output projection with fused softmax normalization
    gemm_out_kernel<<<gridS, 256, 0, stream>>>(agg, zsum, Wo, bo, out, NSRC);
}
